// MultiHeadAttention_47820165874179
// MI455X (gfx1250) — matrix operand order repaired and checked by reading, not compiled, not run on hardware
//
#include <hip/hip_runtime.h>

#define HEADS 16
#define NSEQ  2048
#define EMB   1024
#define DHEAD 64
#define BATCH 2
#define NTOK  (BATCH * NSEQ)   // 4096

typedef __attribute__((ext_vector_type(16))) __bf16 v16bf;
typedef __attribute__((ext_vector_type(8)))  __bf16 v8bf;
typedef __attribute__((ext_vector_type(8)))  float  v8f;

union V16 { v16bf v; v8bf h[2]; };

__device__ inline v8f wmma_bf16(v16bf a, v16bf b, v8f c) {
    return __builtin_amdgcn_wmma_f32_16x16x32_bf16(
        /*neg_a=*/false, a, /*neg_b=*/false, b,
        /*c_mod=*/(short)0, c, /*reuse_a=*/false, /*reuse_b=*/false);
}

// Load one 16x32 bf16 A-fragment for row `arow` at K offset kk.
__device__ inline void load_afrag(const __bf16* __restrict__ arow, int kk,
                                  int kbase, V16& a) {
    const __bf16* p = arow + kk;
    a.h[0] = *(const v8bf*)(p + kbase);
    a.h[1] = *(const v8bf*)(p + kbase + 8);
}

// Load a full 32xK_chunk A pair + 4 B fragments at K offset kk.
__device__ inline void load_tile(const __bf16* __restrict__ arow0,
                                 const __bf16* __restrict__ arow1,
                                 const __bf16* __restrict__ bbase,
                                 int kk, int kbase,
                                 V16& a0, V16& a1, v16bf (&b)[4]) {
    load_afrag(arow0, kk, kbase, a0);
    load_afrag(arow1, kk, kbase, a1);
#pragma unroll
    for (int f = 0; f < 4; ++f)
        b[f] = *(const v16bf*)(bbase + kk + (size_t)f * 16 * EMB);
}

__device__ inline void compute_tile(const V16& a0, const V16& a1,
                                    const v16bf (&b)[4], v8f (&c)[2][4]) {
#pragma unroll
    for (int f = 0; f < 4; ++f) {
        c[0][f] = wmma_bf16(a0.v, b[f], c[0][f]);
        c[1][f] = wmma_bf16(a1.v, b[f], c[1][f]);
    }
}

// ---------------- conversion kernels ----------------

__global__ void conv_bf16_kernel(const float* __restrict__ src,
                                 __bf16* __restrict__ dst, int n) {
    int i = blockIdx.x * blockDim.x + threadIdx.x;
    int stride = gridDim.x * blockDim.x;
    for (; i < n; i += stride) dst[i] = (__bf16)src[i];
}

// src [K][Ncols] (f32, row-major) -> dst [Ncols][K] (bf16)
__global__ void transpose_bf16_kernel(const float* __restrict__ src,
                                      __bf16* __restrict__ dst,
                                      int K, int Ncols) {
    int i = blockIdx.x * blockDim.x + threadIdx.x;
    int total = K * Ncols;
    int stride = gridDim.x * blockDim.x;
    for (; i < total; i += stride) {
        int nn = i / K;
        int kk = i - nn * K;
        dst[i] = (__bf16)src[(size_t)kk * Ncols + nn];
    }
}

// ---------------- QKV GEMM + scatter ----------------
// 32x64 output tile per wave; explicit ping-pong K pipeline (no buffer copies).
// C[tok, c] = xb[tok,:] @ WqkvT[c,:] + bqkv[c], c = h*192 + d*3 + sel
__global__ void __launch_bounds__(32, 1)
qkv_gemm_kernel(const __bf16* __restrict__ xb,
                const __bf16* __restrict__ WqkvT,
                const float* __restrict__ bqkv,
                __bf16* __restrict__ Qb,   // [B,H,N,D]
                __bf16* __restrict__ Kb,   // [B,H,N,D]
                __bf16* __restrict__ Vt) { // [B,H,D,N]
    const int l  = threadIdx.x;
    const int lm = l & 15;
    const int lh = l >> 4;
    const int kbase = lh ? 16 : 0;
    const int mt = blockIdx.x;           // 32-token tile (0..127)
    const int nt = blockIdx.y;           // 64-col tile (0..47)

    const __bf16* arow0 = xb + (size_t)(mt * 32 + lm) * EMB;
    const __bf16* arow1 = arow0 + (size_t)16 * EMB;
    const __bf16* bbase = WqkvT + (size_t)(nt * 64 + lm) * EMB + lh * 16;

    v8f c[2][4] = {};

    V16 a0A, a1A, a0B, a1B;
    v16bf bA[4], bB[4];

    // 32 K-chunks of 32: ping-pong A<->B, no register moves.
    load_tile(arow0, arow1, bbase, 0, kbase, a0A, a1A, bA);
    for (int kc = 0; kc < 30; kc += 2) {
        load_tile(arow0, arow1, bbase, (kc + 1) * 32, kbase, a0B, a1B, bB);
        compute_tile(a0A, a1A, bA, c);
        load_tile(arow0, arow1, bbase, (kc + 2) * 32, kbase, a0A, a1A, bA);
        compute_tile(a0B, a1B, bB, c);
    }
    load_tile(arow0, arow1, bbase, 31 * 32, kbase, a0B, a1B, bB);
    compute_tile(a0A, a1A, bA, c);   // chunk 30
    compute_tile(a0B, a1B, bB, c);   // chunk 31

    const float qscale = 0.03125f * 1.44269504088896f;  // (1/32)*log2(e)
#pragma unroll
    for (int s = 0; s < 2; ++s) {
#pragma unroll
        for (int f = 0; f < 4; ++f) {
            int col = nt * 64 + f * 16 + lm;
            int hh  = col / 192;
            int rem = col - hh * 192;
            int dd  = rem / 3;
            int sel = rem - dd * 3;
            float bias = bqkv[col];
#pragma unroll
            for (int r = 0; r < 8; ++r) {
                int row = lh ? (r + 8) : r;
                int t   = mt * 32 + s * 16 + row;   // global token
                int bI  = t >> 11;                  // / NSEQ
                int n   = t & (NSEQ - 1);
                float val = c[s][f][r] + bias;
                size_t bh = (size_t)(bI * HEADS + hh);
                if (sel == 0)
                    Qb[(bh * NSEQ + n) * DHEAD + dd] = (__bf16)(val * qscale);
                else if (sel == 1)
                    Kb[(bh * NSEQ + n) * DHEAD + dd] = (__bf16)val;
                else
                    Vt[(bh * DHEAD + dd) * NSEQ + n] = (__bf16)val;
            }
        }
    }
}

// ---------------- flash attention: one wave32 per 16-query tile ----------------
__global__ void __launch_bounds__(32, 1)
attn_kernel(const __bf16* __restrict__ Qb,
            const __bf16* __restrict__ Kb,
            const __bf16* __restrict__ Vt,
            __bf16* __restrict__ Ob) {   // [B,N,E] bf16
    const int l  = threadIdx.x;
    const int lm = l & 15;
    const int lh = l >> 4;
    const int kbase = lh ? 16 : 0;
    const int mt = blockIdx.x;    // query tile 0..127
    const int h  = blockIdx.y;
    const int b  = blockIdx.z;
    const size_t bh = (size_t)(b * HEADS + h);

    const __bf16* Qh = Qb + bh * NSEQ * DHEAD;
    const __bf16* Kh = Kb + bh * NSEQ * DHEAD;
    const __bf16* Vh = Vt + bh * DHEAD * NSEQ;

    // Q A-fragments, two K=32 chunks of d=64 (loaded once)
    V16 aQ[2];
    {
        const __bf16* qrow = Qh + (size_t)(mt * 16 + lm) * DHEAD;
        aQ[0].h[0] = *(const v8bf*)(qrow + kbase);
        aQ[0].h[1] = *(const v8bf*)(qrow + kbase + 8);
        aQ[1].h[0] = *(const v8bf*)(qrow + 32 + kbase);
        aQ[1].h[1] = *(const v8bf*)(qrow + 32 + kbase + 8);
    }

    v8f o0 = {}, o1 = {}, o2 = {}, o3 = {};
    float mrun[8], lrun[8];
#pragma unroll
    for (int r = 0; r < 8; ++r) { mrun[r] = -3.0e38f; lrun[r] = 0.f; }

    __shared__ __align__(16) __bf16 Plds[16][40];   // padded: 80B row stride

    for (int j = 0; j < NSEQ / 32; ++j) {
        const __bf16* k0 = Kh + (size_t)(j * 32 + lm) * DHEAD;
        const __bf16* k1 = Kh + (size_t)(j * 32 + 16 + lm) * DHEAD;
        const __bf16* vbase = Vh + (size_t)j * 32 + lh * 16;

        // Issue ALL operand loads for this block up front (K then V);
        // the softmax VALU section below covers the V-load latency.
        v16bf bK0 = *(const v16bf*)(k0 + lh * 16);
        v16bf bK1 = *(const v16bf*)(k0 + 32 + lh * 16);
        v16bf bK2 = *(const v16bf*)(k1 + lh * 16);
        v16bf bK3 = *(const v16bf*)(k1 + 32 + lh * 16);
        v16bf bV0 = *(const v16bf*)(vbase + (size_t)(0 * 16 + lm) * NSEQ);
        v16bf bV1 = *(const v16bf*)(vbase + (size_t)(1 * 16 + lm) * NSEQ);
        v16bf bV2 = *(const v16bf*)(vbase + (size_t)(2 * 16 + lm) * NSEQ);
        v16bf bV3 = *(const v16bf*)(vbase + (size_t)(3 * 16 + lm) * NSEQ);

        // Prefetch next key/value block into cache (global_prefetch_b8).
        if (j + 1 < NSEQ / 32) {
            __builtin_prefetch(k0 + (size_t)32 * DHEAD, 0, 3);
            __builtin_prefetch(k1 + (size_t)32 * DHEAD, 0, 3);
            __builtin_prefetch(vbase + 32, 0, 3);
        }

        // ---- S = Qscaled @ K^T ----
        v8f s0 = {}, s1 = {};
        s0 = wmma_bf16(aQ[0].v, bK0, s0);
        s0 = wmma_bf16(aQ[1].v, bK1, s0);
        s1 = wmma_bf16(aQ[0].v, bK2, s1);
        s1 = wmma_bf16(aQ[1].v, bK3, s1);

        // ---- online softmax (base-2; scale folded into Q) ----
        float alpha[8];
#pragma unroll
        for (int r = 0; r < 8; ++r) {
            float mx = fmaxf(s0[r], s1[r]);
            mx = fmaxf(mx, __shfl_xor(mx, 1, 32));
            mx = fmaxf(mx, __shfl_xor(mx, 2, 32));
            mx = fmaxf(mx, __shfl_xor(mx, 4, 32));
            mx = fmaxf(mx, __shfl_xor(mx, 8, 32));
            float mnew = fmaxf(mrun[r], mx);
            float a  = exp2f(mrun[r] - mnew);
            float p0 = exp2f(s0[r] - mnew);
            float p1 = exp2f(s1[r] - mnew);
            float ps = p0 + p1;
            ps += __shfl_xor(ps, 1, 32);
            ps += __shfl_xor(ps, 2, 32);
            ps += __shfl_xor(ps, 4, 32);
            ps += __shfl_xor(ps, 8, 32);
            lrun[r] = lrun[r] * a + ps;
            mrun[r] = mnew;
            alpha[r] = a;
            s0[r] = p0;
            s1[r] = p1;
        }
#pragma unroll
        for (int r = 0; r < 8; ++r) {
            o0[r] *= alpha[r]; o1[r] *= alpha[r];
            o2[r] *= alpha[r]; o3[r] *= alpha[r];
        }

        // ---- C-layout P -> A-layout via LDS ----
        __syncthreads();
#pragma unroll
        for (int r = 0; r < 8; ++r) {
            int row = lh ? (r + 8) : r;
            Plds[row][lm]      = (__bf16)s0[r];
            Plds[row][lm + 16] = (__bf16)s1[r];
        }
        __syncthreads();
        V16 aP;
        aP.h[0] = *(const v8bf*)(&Plds[lm][kbase]);
        aP.h[1] = *(const v8bf*)(&Plds[lm][kbase + 8]);

        // ---- O += P @ V ----
        o0 = wmma_bf16(aP.v, bV0, o0);
        o1 = wmma_bf16(aP.v, bV1, o1);
        o2 = wmma_bf16(aP.v, bV2, o2);
        o3 = wmma_bf16(aP.v, bV3, o3);
    }

    // ---- finalize and write O in [B,N,E] (e = h*64 + d) ----
#pragma unroll
    for (int r = 0; r < 8; ++r) {
        float inv = 1.0f / lrun[r];
        o0[r] *= inv; o1[r] *= inv; o2[r] *= inv; o3[r] *= inv;
    }
#pragma unroll
    for (int r = 0; r < 8; ++r) {
        int row = lh ? (r + 8) : r;
        size_t tok = (size_t)b * NSEQ + mt * 16 + row;
        __bf16* op = Ob + tok * EMB + h * DHEAD + lm;
        op[0]  = (__bf16)o0[r];
        op[16] = (__bf16)o1[r];
        op[32] = (__bf16)o2[r];
        op[48] = (__bf16)o3[r];
    }
}

// ---------------- output projection: 32x64 tile per wave, ping-pong pipeline ----------------
__global__ void __launch_bounds__(32, 1)
proj_gemm_kernel(const __bf16* __restrict__ Ob,
                 const __bf16* __restrict__ WprojT,
                 const float* __restrict__ bproj,
                 float* __restrict__ out) {
    const int l  = threadIdx.x;
    const int lm = l & 15;
    const int lh = l >> 4;
    const int kbase = lh ? 16 : 0;
    const int mt = blockIdx.x;   // 0..127
    const int nt = blockIdx.y;   // 0..15

    const __bf16* arow0 = Ob + (size_t)(mt * 32 + lm) * EMB;
    const __bf16* arow1 = arow0 + (size_t)16 * EMB;
    const __bf16* bbase = WprojT + (size_t)(nt * 64 + lm) * EMB + lh * 16;

    v8f c[2][4] = {};

    V16 a0A, a1A, a0B, a1B;
    v16bf bA[4], bB[4];

    load_tile(arow0, arow1, bbase, 0, kbase, a0A, a1A, bA);
    for (int kc = 0; kc < 30; kc += 2) {
        load_tile(arow0, arow1, bbase, (kc + 1) * 32, kbase, a0B, a1B, bB);
        compute_tile(a0A, a1A, bA, c);
        load_tile(arow0, arow1, bbase, (kc + 2) * 32, kbase, a0A, a1A, bA);
        compute_tile(a0B, a1B, bB, c);
    }
    load_tile(arow0, arow1, bbase, 31 * 32, kbase, a0B, a1B, bB);
    compute_tile(a0A, a1A, bA, c);   // chunk 30
    compute_tile(a0B, a1B, bB, c);   // chunk 31

#pragma unroll
    for (int s = 0; s < 2; ++s) {
#pragma unroll
        for (int f = 0; f < 4; ++f) {
            int col = nt * 64 + f * 16 + lm;
            float bias = bproj[col];
#pragma unroll
            for (int r = 0; r < 8; ++r) {
                int row = lh ? (r + 8) : r;
                out[(size_t)(mt * 32 + s * 16 + row) * EMB + col] = c[s][f][r] + bias;
            }
        }
    }
}

// ---------------- launch ----------------
extern "C" void kernel_launch(void* const* d_in, const int* in_sizes, int n_in,
                              void* d_out, int out_size, void* d_ws, size_t ws_size,
                              hipStream_t stream) {
    const float* x     = (const float*)d_in[0];
    const float* Wqkv  = (const float*)d_in[1];
    const float* bqkv  = (const float*)d_in[2];
    const float* Wproj = (const float*)d_in[3];
    const float* bproj = (const float*)d_in[4];
    float* out = (float*)d_out;

    char* ws = (char*)d_ws;
    __bf16* xb     = (__bf16*)ws;  ws += (size_t)NTOK * EMB * 2;          // 8 MB
    __bf16* WqkvT  = (__bf16*)ws;  ws += (size_t)3 * EMB * EMB * 2;       // 6 MB
    __bf16* WprojT = (__bf16*)ws;  ws += (size_t)EMB * EMB * 2;           // 2 MB
    __bf16* Qb     = (__bf16*)ws;  ws += (size_t)NTOK * EMB * 2;          // 8 MB
    __bf16* Kb     = (__bf16*)ws;  ws += (size_t)NTOK * EMB * 2;          // 8 MB
    __bf16* Vt     = (__bf16*)ws;  ws += (size_t)NTOK * EMB * 2;          // 8 MB
    __bf16* Ob     = (__bf16*)ws;  ws += (size_t)NTOK * EMB * 2;          // 8 MB

    conv_bf16_kernel<<<1024, 256, 0, stream>>>(x, xb, NTOK * EMB);
    transpose_bf16_kernel<<<1024, 256, 0, stream>>>(Wqkv, WqkvT, EMB, 3 * EMB);
    transpose_bf16_kernel<<<1024, 256, 0, stream>>>(Wproj, WprojT, EMB, EMB);

    qkv_gemm_kernel<<<dim3(NTOK / 32, (3 * EMB) / 64), 32, 0, stream>>>(
        xb, WqkvT, bqkv, Qb, Kb, Vt);

    attn_kernel<<<dim3(NSEQ / 16, HEADS, BATCH), 32, 0, stream>>>(Qb, Kb, Vt, Ob);

    proj_gemm_kernel<<<dim3(NTOK / 32, EMB / 64), 32, 0, stream>>>(
        Ob, WprojT, bproj, out);
}
